// FFTNet_72310069395648
// MI455X (gfx1250) — matrix operand order repaired and checked by reading, not compiled, not run on hardware
//
#include <hip/hip_runtime.h>

typedef __attribute__((ext_vector_type(16))) __bf16 bf16x16;
typedef __attribute__((ext_vector_type(8)))  float  f32x8;
typedef __attribute__((ext_vector_type(4)))  unsigned int u32x4;

#define RF     2048
#define TLEN   16384
#define CH     256
#define NCIN   80
#define NB     4
#define L0     (RF + TLEN)     // 18432
#define LCAP   18560           // L0 + 128 pad (tail tiles read/write garbage here only)
#define LCOND  17408           // L0 - RF/2
#define LDSROW 264             // 256 + 8 bf16 pad -> 528B row, 16B aligned, 4-bank skew
#define NTILE  128             // time columns per block

union FragU { u32x4 q[2]; bf16x16 v; };

__device__ __forceinline__ float bf2f(unsigned short h) {
    unsigned int u = ((unsigned int)h) << 16;
    float f; __builtin_memcpy(&f, &u, 4); return f;
}
__device__ __forceinline__ unsigned short f2bf(float f) {
    unsigned int u; __builtin_memcpy(&u, &f, 4);
    u += 0x7FFFu + ((u >> 16) & 1u);            // round-to-nearest-even
    return (unsigned short)(u >> 16);
}

// A fragment: W row-major [CH][CH] bf16, rows mBase..mBase+15, K = k0..k0+31.
// Contiguous k order used here (same as B): lanes 0-15 hold K 0-15; lanes 16-31 hold K 16-31.
__device__ __forceinline__ bf16x16 load_a_frag(const unsigned short* __restrict__ W,
                                               int mBase, int k0, int lane) {
    int m  = mBase + (lane & 15);
    int hi = lane >> 4;
    const unsigned short* p = W + m * CH + k0 + hi * 16;
    FragU u;
    u.q[0] = *(const u32x4*)(p);       // K +0..7 of this lane group
    u.q[1] = *(const u32x4*)(p + 8);   // K +8..15 of this lane group
    return u.v;
}

// B fragment: H time-major [t][CH] bf16, tile cols tBase..tBase+15, K = k0..k0+31.
// Contiguous k order used here (same as A): lanes 0-15 (n=lane) hold K k0..k0+15; lanes 16-31 hold K k0+16..k0+31.
__device__ __forceinline__ bf16x16 load_b_frag(const unsigned short* __restrict__ H,
                                               int tBase, int k0, int lane) {
    int n  = lane & 15;
    int kb = k0 + (lane >> 4) * 16;
    const unsigned short* p = H + (size_t)(tBase + n) * CH + kb;
    FragU u;
    u.q[0] = *(const u32x4*)(p);
    u.q[1] = *(const u32x4*)(p + 8);
    return u.v;
}

__device__ __forceinline__ bf16x16 load_b_frag_lds(const unsigned short* lds,
                                                   int tBase, int k0, int lane) {
    int n  = lane & 15;
    int kb = k0 + (lane >> 4) * 16;
    const unsigned short* p = lds + (tBase + n) * LDSROW + kb;
    FragU u;
    u.q[0] = *(const u32x4*)(p);
    u.q[1] = *(const u32x4*)(p + 8);
    return u.v;
}

#define WMMA_BF16(a, b, c) \
    __builtin_amdgcn_wmma_f32_16x16x32_bf16(false, (a), false, (b), (short)0, (c), false, false)

// ---------------- prep kernels ----------------

__global__ void pack_bf16(const float* __restrict__ src, unsigned short* __restrict__ dst, int n) {
    int i = blockIdx.x * 256 + threadIdx.x;
    if (i < n) dst[i] = f2bf(src[i]);
}

// h0[b][t][c] = (t < RF) ? (c==127 ? 1 : 0) : x[b][c][t-RF]
__global__ void build_h0(const float* __restrict__ x, unsigned short* __restrict__ h0) {
    int b = blockIdx.y;
    int t = blockIdx.x * 256 + threadIdx.x;          // t < L0 (grid exact)
    unsigned int* d32 = (unsigned int*)(h0 + ((size_t)b * LCAP + t) * CH);
    if (t < RF) {
        #pragma unroll 8
        for (int i = 0; i < CH / 2; i++)
            d32[i] = (i == 63) ? 0x3F800000u : 0u;   // c=127 high half = bf16(1.0)
    } else {
        const float* xs = x + (size_t)b * CH * TLEN + (t - RF);
        #pragma unroll 4
        for (int i = 0; i < CH / 2; i++) {
            unsigned int lo = f2bf(xs[(size_t)(2 * i)     * TLEN]);
            unsigned int hi = f2bf(xs[(size_t)(2 * i + 1) * TLEN]);
            d32[i] = lo | (hi << 16);
        }
    }
}

// cond[b][t][o] = sum_k Wcl[o][k]*cpad[k][t] + Wcr[o][k]*cpad[k][t+RF/2] + bcl[o] + bcr[o]
__global__ void cond_kernel(const float* __restrict__ c,
                            const float* __restrict__ Wcl, const float* __restrict__ bcl,
                            const float* __restrict__ Wcr, const float* __restrict__ bcr,
                            unsigned short* __restrict__ cond) {
    int b = blockIdx.z;
    int o = blockIdx.y;
    int t = blockIdx.x * 256 + threadIdx.x;          // t < LCOND (grid exact)
    float acc = bcl[o] + bcr[o];
    const float* cb = c + (size_t)b * NCIN * TLEN;
    int tl = t - RF;          // cpad[t]        -> c[t-RF]
    int tr = t - RF / 2;      // cpad[t+RF/2]   -> c[t-RF/2]
    #pragma unroll 4
    for (int k = 0; k < NCIN; k++) {
        if (tl >= 0) acc += Wcl[o * NCIN + k] * cb[(size_t)k * TLEN + tl];
        if (tr >= 0) acc += Wcr[o * NCIN + k] * cb[(size_t)k * TLEN + tr];
    }
    cond[((size_t)b * LCOND + t) * CH + o] = f2bf(acc);
}

// ---------------- fused layer kernel ----------------
// Block: 256 thr (8 waves = 4(M) x 2(N)). Block tile: 128 time x 256 channels.
// Wave tile: 64(M) x 64(N) = 16 WMMA accumulators.
// Stage1: relu(Wl*h[t] + Wr*h[t+shift] + bl + br (+cond)) -> LDS bf16
// Stage2: relu(Wp*z + bp) -> hout [b][t][c]

__global__ __launch_bounds__(256) void fftnet_layer(
    const unsigned short* __restrict__ Wl, const unsigned short* __restrict__ Wr,
    const float* __restrict__ bl, const float* __restrict__ br,
    const unsigned short* __restrict__ Wp, const float* __restrict__ bp,
    const unsigned short* __restrict__ cond,      // may be null
    const unsigned short* __restrict__ hin,
    unsigned short* __restrict__ hout,
    int shift)
{
    __shared__ unsigned short lds[NTILE * LDSROW];
    const int b    = blockIdx.y;
    const int t0   = blockIdx.x * NTILE;
    const int lane = threadIdx.x & 31;
    const int wave = threadIdx.x >> 5;
    const int mB   = (wave & 3) * 64;     // 4 m-tiles of 16
    const int nB   = (wave >> 2) * 64;    // 4 n-tiles of 16

    const unsigned short* hbL = hin + (size_t)b * LCAP * CH + (size_t)t0 * CH;
    const unsigned short* hbR = hbL + (size_t)shift * CH;

    const f32x8 fz = {0.f, 0.f, 0.f, 0.f, 0.f, 0.f, 0.f, 0.f};
    f32x8 acc[4][4];
    #pragma unroll
    for (int i = 0; i < 4; i++)
        #pragma unroll
        for (int j = 0; j < 4; j++) acc[i][j] = fz;

    for (int k0 = 0; k0 < CH; k0 += 32) {
        bf16x16 a[4];
        #pragma unroll
        for (int mt = 0; mt < 4; mt++) a[mt] = load_a_frag(Wl, mB + mt * 16, k0, lane);
        #pragma unroll
        for (int nt = 0; nt < 4; nt++) {
            bf16x16 bb = load_b_frag(hbL, nB + nt * 16, k0, lane);
            #pragma unroll
            for (int mt = 0; mt < 4; mt++)
                acc[mt][nt] = WMMA_BF16(a[mt], bb, acc[mt][nt]);
        }
        #pragma unroll
        for (int mt = 0; mt < 4; mt++) a[mt] = load_a_frag(Wr, mB + mt * 16, k0, lane);
        #pragma unroll
        for (int nt = 0; nt < 4; nt++) {
            bf16x16 bb = load_b_frag(hbR, nB + nt * 16, k0, lane);
            #pragma unroll
            for (int mt = 0; mt < 4; mt++)
                acc[mt][nt] = WMMA_BF16(a[mt], bb, acc[mt][nt]);
        }
    }

    const int hi = lane >> 4;
    const int n  = lane & 15;
    #pragma unroll
    for (int mt = 0; mt < 4; mt++) {
        const int mrow = mB + mt * 16 + hi * 8;
        float bias[8];
        #pragma unroll
        for (int r = 0; r < 8; r++) bias[r] = bl[mrow + r] + br[mrow + r];
        #pragma unroll
        for (int nt = 0; nt < 4; nt++) {
            const int tl = nB + nt * 16 + n;
            f32x8 v = acc[mt][nt];
            if (cond) {
                const unsigned short* cp = cond + ((size_t)b * LCOND + (t0 + tl)) * CH + mrow;
                u32x4 cq = *(const u32x4*)cp;
                const unsigned short* cs = (const unsigned short*)&cq;
                #pragma unroll
                for (int r = 0; r < 8; r++) v[r] += bf2f(cs[r]);
            }
            union { unsigned short h[8]; u32x4 q; } st;
            #pragma unroll
            for (int r = 0; r < 8; r++) {
                float z = v[r] + bias[r];
                st.h[r] = f2bf(z > 0.f ? z : 0.f);
            }
            *(u32x4*)&lds[tl * LDSROW + mrow] = st.q;
        }
    }
    __syncthreads();

    #pragma unroll
    for (int i = 0; i < 4; i++)
        #pragma unroll
        for (int j = 0; j < 4; j++) acc[i][j] = fz;

    for (int k0 = 0; k0 < CH; k0 += 32) {
        bf16x16 a[4];
        #pragma unroll
        for (int mt = 0; mt < 4; mt++) a[mt] = load_a_frag(Wp, mB + mt * 16, k0, lane);
        #pragma unroll
        for (int nt = 0; nt < 4; nt++) {
            bf16x16 bb = load_b_frag_lds(lds, nB + nt * 16, k0, lane);
            #pragma unroll
            for (int mt = 0; mt < 4; mt++)
                acc[mt][nt] = WMMA_BF16(a[mt], bb, acc[mt][nt]);
        }
    }

    unsigned short* ho = hout + (size_t)b * LCAP * CH + (size_t)t0 * CH;
    #pragma unroll
    for (int mt = 0; mt < 4; mt++) {
        const int mrow = mB + mt * 16 + hi * 8;
        #pragma unroll
        for (int nt = 0; nt < 4; nt++) {
            const int tl = nB + nt * 16 + n;
            f32x8 v = acc[mt][nt];
            union { unsigned short h[8]; u32x4 q; } st;
            #pragma unroll
            for (int r = 0; r < 8; r++) {
                float z = v[r] + bp[mrow + r];
                st.h[r] = f2bf(z > 0.f ? z : 0.f);
            }
            *(u32x4*)&ho[(size_t)tl * CH + mrow] = st.q;
        }
    }
}

// Final: out[b][o][t] = Wlast*h + blast, f32
__global__ __launch_bounds__(256) void fftnet_last(
    const unsigned short* __restrict__ Wlast, const float* __restrict__ blast,
    const unsigned short* __restrict__ hin, float* __restrict__ out)
{
    const int b    = blockIdx.y;
    const int t0   = blockIdx.x * NTILE;
    const int lane = threadIdx.x & 31;
    const int wave = threadIdx.x >> 5;
    const int mB   = (wave & 3) * 64;
    const int nB   = (wave >> 2) * 64;
    const unsigned short* hb = hin + (size_t)b * LCAP * CH + (size_t)t0 * CH;

    const f32x8 fz = {0.f, 0.f, 0.f, 0.f, 0.f, 0.f, 0.f, 0.f};
    f32x8 acc[4][4];
    #pragma unroll
    for (int i = 0; i < 4; i++)
        #pragma unroll
        for (int j = 0; j < 4; j++) acc[i][j] = fz;

    for (int k0 = 0; k0 < CH; k0 += 32) {
        bf16x16 a[4];
        #pragma unroll
        for (int mt = 0; mt < 4; mt++) a[mt] = load_a_frag(Wlast, mB + mt * 16, k0, lane);
        #pragma unroll
        for (int nt = 0; nt < 4; nt++) {
            bf16x16 bb = load_b_frag(hb, nB + nt * 16, k0, lane);
            #pragma unroll
            for (int mt = 0; mt < 4; mt++)
                acc[mt][nt] = WMMA_BF16(a[mt], bb, acc[mt][nt]);
        }
    }

    const int hi = lane >> 4;
    const int n  = lane & 15;
    #pragma unroll
    for (int mt = 0; mt < 4; mt++) {
        const int mrow = mB + mt * 16 + hi * 8;
        #pragma unroll
        for (int nt = 0; nt < 4; nt++) {
            const int t = t0 + nB + nt * 16 + n;
            f32x8 v = acc[mt][nt];
            #pragma unroll
            for (int r = 0; r < 8; r++)
                out[((size_t)b * CH + (mrow + r)) * TLEN + t] = v[r] + blast[mrow + r];
        }
    }
}

// ---------------- launcher ----------------

extern "C" void kernel_launch(void* const* d_in, const int* in_sizes, int n_in,
                              void* d_out, int out_size, void* d_ws, size_t ws_size,
                              hipStream_t stream) {
    (void)in_sizes; (void)n_in; (void)out_size; (void)ws_size;
    const float* x     = (const float*)d_in[0];
    const float* c     = (const float*)d_in[1];
    const float* Wl    = (const float*)d_in[2];
    const float* bl    = (const float*)d_in[3];
    const float* Wr    = (const float*)d_in[4];
    const float* br    = (const float*)d_in[5];
    const float* Wp    = (const float*)d_in[6];
    const float* bp    = (const float*)d_in[7];
    const float* Wcl   = (const float*)d_in[8];
    const float* bcl   = (const float*)d_in[9];
    const float* Wcr   = (const float*)d_in[10];
    const float* bcr   = (const float*)d_in[11];
    const float* Wlast = (const float*)d_in[12];
    const float* blast = (const float*)d_in[13];
    float* out = (float*)d_out;

    unsigned short* ws = (unsigned short*)d_ws;
    size_t off = 0;
    unsigned short* hA     = ws + off; off += (size_t)NB * LCAP  * CH;
    unsigned short* hBuf   = ws + off; off += (size_t)NB * LCAP  * CH;
    unsigned short* condb  = ws + off; off += (size_t)NB * LCOND * CH;
    unsigned short* wlb    = ws + off; off += (size_t)11 * CH * CH;
    unsigned short* wrb    = ws + off; off += (size_t)11 * CH * CH;
    unsigned short* wpb    = ws + off; off += (size_t)11 * CH * CH;
    unsigned short* wlastb = ws + off; off += (size_t)CH * CH;

    {
        int n = 11 * CH * CH;
        dim3 g((n + 255) / 256);
        pack_bf16<<<g, 256, 0, stream>>>(Wl, wlb, n);
        pack_bf16<<<g, 256, 0, stream>>>(Wr, wrb, n);
        pack_bf16<<<g, 256, 0, stream>>>(Wp, wpb, n);
        int m = CH * CH;
        pack_bf16<<<dim3((m + 255) / 256), 256, 0, stream>>>(Wlast, wlastb, m);
    }
    build_h0<<<dim3(L0 / 256, NB), 256, 0, stream>>>(x, hA);
    cond_kernel<<<dim3(LCOND / 256, CH, NB), 256, 0, stream>>>(c, Wcl, bcl, Wcr, bcr, condb);

    unsigned short* cur = hA;
    unsigned short* nxt = hBuf;
    int L = L0;
    for (int i = 0; i < 11; i++) {
        int shift = 1 << (10 - i);          // RF >> (i+1)
        int Lout = L - shift;
        dim3 grid((Lout + NTILE - 1) / NTILE, NB);
        fftnet_layer<<<grid, 256, 0, stream>>>(
            wlb + (size_t)i * CH * CH, wrb + (size_t)i * CH * CH,
            bl + i * CH, br + i * CH,
            wpb + (size_t)i * CH * CH, bp + i * CH,
            (i == 0) ? condb : (const unsigned short*)nullptr,
            cur, nxt, shift);
        unsigned short* t = cur; cur = nxt; nxt = t;
        L = Lout;
    }
    fftnet_last<<<dim3(TLEN / NTILE, NB), 256, 0, stream>>>(wlastb, blast, cur, out);
}
